// SelfAttention_85581518340629
// MI455X (gfx1250) — matrix operand order repaired and checked by reading, not compiled, not run on hardware
//
#include <hip/hip_runtime.h>
#include <hip/hip_bf16.h>
#include <math.h>

#define DIM    1024
#define HEADS  16
#define DHEAD  64
#define EPSV   1e-6f
#define NIT    (DIM / 32)          // K-steps of 32

typedef __attribute__((ext_vector_type(16))) _Float16 v16h;
typedef __attribute__((ext_vector_type(8)))  _Float16 v8h;
typedef __attribute__((ext_vector_type(8)))  float    v8f;

// LDS staging strides (f16 elements). 40 f16 = 80B rows: 16B-aligned vector
// accesses, 20-dword row stride -> conflict-free b128 reads across 16 lanes.
#define AS_STRIDE 40                       // A tile: 128 rows x 32 K (+8 pad)
#define BT_STRIDE 40                       // B tile (transposed): 64 cols x 32 K (+8 pad)
#define AS_BYTES  (128 * AS_STRIDE * 2)    // 10240
#define BT_BYTES  (64  * BT_STRIDE * 2)    // 5120
#define STAGE_BYTES (AS_BYTES + BT_BYTES)  // 15360 per buffer (double buffered)
#define P_STRIDE  65                       // f32 epilogue tile stride
#define SMEM_BYTES (128 * P_STRIDE * 4)    // 33280 >= 2*STAGE_BYTES (30720)

__device__ __forceinline__ v16h ld_afrag(const _Float16* p) {
    // A operand: lane holds K {kb..kb+7} in VGPRs 0-3 and {16+kb..23+kb} in 4-7
    v8h lo = *(const v8h*)p;
    v8h hi = *(const v8h*)(p + 16);
    return __builtin_shufflevector(lo, hi, 0,1,2,3,4,5,6,7,8,9,10,11,12,13,14,15);
}
__device__ __forceinline__ v16h ld_bfrag(const _Float16* p) {
    // B operand: lane holds K {kb..kb+7} in VGPRs 0-3 and {16+kb..23+kb} in 4-7
    v8h lo = *(const v8h*)p;
    v8h hi = *(const v8h*)(p + 16);
    return __builtin_shufflevector(lo, hi, 0,1,2,3,4,5,6,7,8,9,10,11,12,13,14,15);
}

// ---------------------------------------------------------------------------
// GEMM mainloop, 128x64 tile per 128-thread block (4 waves).
// Wave w owns rows [32w, 32w+32): 2 A fragments x 4 shared B fragments
// -> 8 v_wmma_f32_16x16x32_f16 per K-step. LDS double-buffered: one barrier
// per K-step; next tile's global loads overlap current tile's WMMAs.
// ---------------------------------------------------------------------------
__device__ __forceinline__ void gemm_core(const float* __restrict__ A,
                                          const float* __restrict__ B,
                                          char* smem, int rowBase, int colBase,
                                          v8f acc[2][4])
{
    const int tid  = threadIdx.x;
    const int lane = tid & 31;
    const int wave = tid >> 5;

    // WMMA fragment addressing (ISA 16-bit A 16x32 / B 32x16 layouts)
    const int m   = lane & 15;
    const int kbA = (lane >> 4) * 8;
    const int kbB = (lane >> 4) * 8;

    // staging ownership: A -> thread t owns row t (32 consecutive K f32);
    // B -> thread t owns column (t&63), 16 K values (lane-consecutive n => coalesced)
    const int bn  = tid & 63;
    const int bkb = (tid >> 6) * 16;
    const float* aSrc = A + (rowBase + tid) * DIM;
    const float* bSrc = B + colBase + bn;

    auto stage = [&](int buf, int k0) {
        _Float16* As = (_Float16*)(smem + buf * STAGE_BYTES);
        _Float16* Bt = (_Float16*)(smem + buf * STAGE_BYTES + AS_BYTES);
        // ---- A: one row, 32 consecutive f32 -> 32 f16, 4x ds_store_b128
        float4 fa[8];
#pragma unroll
        for (int j = 0; j < 8; ++j) fa[j] = ((const float4*)(aSrc + k0))[j];
        _Float16 ha[32];
#pragma unroll
        for (int j = 0; j < 8; ++j) {
            ha[4*j+0] = (_Float16)fa[j].x; ha[4*j+1] = (_Float16)fa[j].y;
            ha[4*j+2] = (_Float16)fa[j].z; ha[4*j+3] = (_Float16)fa[j].w;
        }
#pragma unroll
        for (int s = 0; s < 4; ++s)
            *(v8h*)(As + tid * AS_STRIDE + 8 * s) = *(const v8h*)(ha + 8 * s);
        // ---- B: one column, 16 K values (DIM-strided loads, wave-coalesced),
        //         packed contiguous along K -> 2x ds_store_b128
        const float* bs = bSrc + (k0 + bkb) * DIM;
        float fb[16];
#pragma unroll
        for (int j = 0; j < 16; ++j) fb[j] = bs[j * DIM];
        _Float16 hb[16];
#pragma unroll
        for (int j = 0; j < 16; ++j) hb[j] = (_Float16)fb[j];
#pragma unroll
        for (int s = 0; s < 2; ++s)
            *(v8h*)(Bt + bn * BT_STRIDE + bkb + 8 * s) = *(const v8h*)(hb + 8 * s);
    };

    auto compute = [&](int buf) {
        const _Float16* As = (const _Float16*)(smem + buf * STAGE_BYTES);
        const _Float16* Bt = (const _Float16*)(smem + buf * STAGE_BYTES + AS_BYTES);
        const _Float16* ar0 = As + (wave * 32 + m) * AS_STRIDE;
        v16h a0 = ld_afrag(ar0 + kbA);
        v16h a1 = ld_afrag(ar0 + 16 * AS_STRIDE + kbA);
#pragma unroll
        for (int g = 0; g < 4; ++g) {
            v16h b = ld_bfrag(Bt + (g * 16 + m) * BT_STRIDE + kbB);
            acc[0][g] = __builtin_amdgcn_wmma_f32_16x16x32_f16(
                false, a0, false, b, (short)0, acc[0][g], false, false);
            acc[1][g] = __builtin_amdgcn_wmma_f32_16x16x32_f16(
                false, a1, false, b, (short)0, acc[1][g], false, false);
        }
    };

    stage(0, 0);
    for (int it = 0; it < NIT; ++it) {
        __syncthreads();                       // buf[it&1] visible to all waves
        if (it + 1 < NIT) stage((it + 1) & 1, (it + 1) * 32);
        compute(it & 1);
    }
}

// ---------------------------------------------------------------------------
// Fused QKV: blockIdx.z selects {0:q (softmax/head), 1:k (elu+1), 2:v (plain)}
// ---------------------------------------------------------------------------
__global__ __launch_bounds__(128) void qkv_gemm_kernel(
    const float* __restrict__ x,
    const float* __restrict__ Wq, const float* __restrict__ Wk, const float* __restrict__ Wv,
    float* __restrict__ qo, float* __restrict__ ko, float* __restrict__ vo)
{
    __shared__ __align__(32) char smem[SMEM_BYTES];
    const int which = blockIdx.z;
    const float* B = (which == 0) ? Wq : (which == 1) ? Wk : Wv;
    float*       C = (which == 0) ? qo : (which == 1) ? ko : vo;
    const int rowBase = blockIdx.y * 128;
    const int colBase = blockIdx.x * 64;       // one head per 64-col tile

    v8f acc[2][4] = {};
    gemm_core(x, B, smem, rowBase, colBase, acc);

    const int tid  = threadIdx.x;
    const int lane = tid & 31;
    const int wave = tid >> 5;
    const int mOff = (lane >> 4) * 8;          // C/D layout: VGPR r -> M = r + mOff
    const int n    = lane & 15;

    if (which == 0) {
        // softmax over the 64 columns (= one head) of each row, via LDS tile
        float* Pt = reinterpret_cast<float*>(smem);   // aliases staging buffers
        __syncthreads();
#pragma unroll
        for (int h = 0; h < 2; ++h)
#pragma unroll
            for (int g = 0; g < 4; ++g)
#pragma unroll
                for (int r = 0; r < 8; ++r)
                    Pt[(wave * 32 + h * 16 + mOff + r) * P_STRIDE + g * 16 + n] = acc[h][g][r];
        __syncthreads();
        {
            const float* row = Pt + tid * P_STRIDE;   // 128 threads -> 128 rows
            float mx = row[0];
#pragma unroll
            for (int j = 1; j < 64; ++j) mx = fmaxf(mx, row[j]);
            float s = 0.f;
#pragma unroll
            for (int j = 0; j < 64; ++j) s += __expf(row[j] - mx);
            const float inv = 1.f / s;
            float* out = C + (rowBase + tid) * DIM + colBase;
#pragma unroll
            for (int j = 0; j < 64; ++j) out[j] = __expf(row[j] - mx) * inv;
        }
    } else {
#pragma unroll
        for (int h = 0; h < 2; ++h)
#pragma unroll
            for (int g = 0; g < 4; ++g)
#pragma unroll
                for (int r = 0; r < 8; ++r) {
                    float val = acc[h][g][r];
                    if (which == 1)   // elu(x)+1 = x+1 (x>=0), e^x (x<0)
                        val = (val > 0.f) ? (val + 1.f) : __expf(val);
                    C[(rowBase + wave * 32 + h * 16 + mOff + r) * DIM +
                      colBase + g * 16 + n] = val;
                }
    }
}

// ---------------------------------------------------------------------------
// Per-position head mixing (head-axis cumsums collapse to a tiny 16x16 causal
// attention over heads per position):
//   attn[h,e] = sum_{h'<=h} S[h,h'] * v[h',e],
//   S[h,h'] = sum_d (q[h,d]/(kcum[h,d]+eps)) * k[h',d]
// ~67 MFLOP total -> plain VALU, one block per position.
// ---------------------------------------------------------------------------
__global__ __launch_bounds__(256) void head_mix_kernel(
    const float* __restrict__ q, const float* __restrict__ k,
    const float* __restrict__ v, float* __restrict__ attn)
{
    __shared__ float sq[DIM];       // q, then overwritten by qn = q/(kcum+eps)
    __shared__ float sk[DIM];
    __shared__ float sv[DIM];
    __shared__ float sS[HEADS][HEADS + 1];

    const int npos = blockIdx.x;
    const int tid  = threadIdx.x;
    const float* qr = q + npos * DIM;
    const float* kr = k + npos * DIM;
    const float* vr = v + npos * DIM;

    for (int i = tid; i < DIM; i += 256) { sq[i] = qr[i]; sk[i] = kr[i]; sv[i] = vr[i]; }
    __syncthreads();

    if (tid < DHEAD) {              // running head-cumsum of k, fold into q
        float run = 0.f;
#pragma unroll
        for (int h = 0; h < HEADS; ++h) {
            run += sk[h * DHEAD + tid];
            sq[h * DHEAD + tid] /= (run + EPSV);
        }
    }
    __syncthreads();

    {                               // S (16x16, causal over heads)
        const int h = tid >> 4, hp = tid & 15;
        float s = 0.f;
        if (hp <= h)
#pragma unroll
            for (int d = 0; d < DHEAD; ++d) s += sq[h * DHEAD + d] * sk[hp * DHEAD + d];
        if (tid < HEADS * HEADS) sS[h][hp] = (hp <= h) ? s : 0.f;
    }
    __syncthreads();

    float* ar = attn + npos * DIM;
    for (int i = tid; i < DIM; i += 256) {
        const int h = i >> 6, e = i & 63;
        float s = 0.f;
#pragma unroll
        for (int hp = 0; hp < HEADS; ++hp) s += sS[h][hp] * sv[hp * DHEAD + e];
        ar[i] = s;
    }
}

// ---------------------------------------------------------------------------
// Output projection: out = attn @ Wo + bo
// ---------------------------------------------------------------------------
__global__ __launch_bounds__(128) void out_proj_kernel(
    const float* __restrict__ attn, const float* __restrict__ Wo,
    const float* __restrict__ bo, float* __restrict__ out)
{
    __shared__ __align__(32) char smem[SMEM_BYTES];
    const int rowBase = blockIdx.y * 128;
    const int colBase = blockIdx.x * 64;

    v8f acc[2][4] = {};
    gemm_core(attn, Wo, smem, rowBase, colBase, acc);

    const int lane = threadIdx.x & 31;
    const int wave = threadIdx.x >> 5;
    const int mOff = (lane >> 4) * 8;
    const int n    = lane & 15;
#pragma unroll
    for (int h = 0; h < 2; ++h)
#pragma unroll
        for (int g = 0; g < 4; ++g)
#pragma unroll
            for (int r = 0; r < 8; ++r) {
                const int col = colBase + g * 16 + n;
                out[(rowBase + wave * 32 + h * 16 + mOff + r) * DIM + col] =
                    acc[h][g][r] + bo[col];
            }
}

extern "C" void kernel_launch(void* const* d_in, const int* in_sizes, int n_in,
                              void* d_out, int out_size, void* d_ws, size_t ws_size,
                              hipStream_t stream) {
    (void)in_sizes; (void)n_in; (void)out_size; (void)ws_size;
    const float* x  = (const float*)d_in[0];
    const float* Wq = (const float*)d_in[1];
    const float* Wk = (const float*)d_in[2];
    const float* Wv = (const float*)d_in[3];
    const float* Wo = (const float*)d_in[4];
    const float* bo = (const float*)d_in[5];
    float* out = (float*)d_out;

    float* qws = (float*)d_ws;                 // 4 MB each, 16 MB total
    float* kws = qws + DIM * DIM;
    float* vws = kws + DIM * DIM;
    float* aws = vws + DIM * DIM;

    dim3 gridQKV(DIM / 64, DIM / 128, 3);      // 16 x 8 x 3 = 384 blocks
    dim3 gridP  (DIM / 64, DIM / 128, 1);

    qkv_gemm_kernel<<<gridQKV, dim3(128), 0, stream>>>(x, Wq, Wk, Wv, qws, kws, vws);
    head_mix_kernel<<<dim3(DIM), dim3(256), 0, stream>>>(qws, kws, vws, aws);
    out_proj_kernel<<<gridP, dim3(128), 0, stream>>>(aws, Wo, bo, out);
}
